// LakeFMModule_12489764897043
// MI455X (gfx1250) — matrix operand order repaired and checked by reading, not compiled, not run on hardware
//
#include <hip/hip_runtime.h>

// ---------------------------------------------------------------------------
// LakeFM transformer forward for gfx1250 (MI455X): bf16 WMMA everywhere.
// ---------------------------------------------------------------------------

typedef __bf16 bf16;
typedef bf16  bf16x8  __attribute__((ext_vector_type(8)));
typedef bf16  bf16x16 __attribute__((ext_vector_type(16)));
typedef float v8f     __attribute__((ext_vector_type(8)));

constexpr int   B_   = 4;
constexpr int   L_   = 1024;
constexpr int   D_   = 1024;
constexpr int   H_   = 16;
constexpr int   NL_  = 4;
constexpr int   Dh_  = D_ / H_;   // 64
constexpr int   DFF_ = 4 * D_;    // 4096
constexpr int   M_   = B_ * L_;   // 4096 tokens
constexpr float EPS_ = 1e-5f;

#define DEV __device__ __forceinline__

// CDNA5 async global->LDS staging (ASYNCcnt path), if the toolchain has it.
// Builtin prototype (probe-discovered via round-2 diagnostic):
//   void __builtin_amdgcn_global_load_async_to_lds_b128(
//       int4 addrspace(1)* gsrc, int4 addrspace(3)* ldst, imm offset, imm cpol)
#if defined(__gfx1250__) &&                                          \
    __has_builtin(__builtin_amdgcn_global_load_async_to_lds_b128) && \
    __has_builtin(__builtin_amdgcn_s_wait_asynccnt)
#define USE_ASYNC_LDS 1
typedef int v4i_ __attribute__((vector_size(16)));
typedef __attribute__((address_space(1))) v4i_ gv4i;
typedef __attribute__((address_space(3))) v4i_ lv4i;
#else
#define USE_ASYNC_LDS 0
#endif

// A-matrix fragment (16x32 bf16, ISA 7.12.2): lane half already folded into p.
// lanes 0-15: K{kb..kb+7, kb+16..kb+23}; lanes 16-31: kb=8 -> K{8-15,24-31}.
DEV bf16x16 fragA(const bf16* p) {
  bf16x8 lo = *(const bf16x8*)p;
  bf16x8 hi = *(const bf16x8*)(p + 16);
  return __builtin_shufflevector(lo, hi, 0,1,2,3,4,5,6,7,8,9,10,11,12,13,14,15);
}

// B-matrix fragment (32x16 bf16): lane n = column n%16, holds K{kb..kb+7,
// kb+16..kb+23} with kb = (lane<16 ? 0 : 8) -- caller folds that into p.
DEV bf16x16 fragB(const bf16* p) {
  bf16x8 lo = *(const bf16x8*)p;
  bf16x8 hi = *(const bf16x8*)(p + 16);
  return __builtin_shufflevector(lo, hi, 0,1,2,3,4,5,6,7,8,9,10,11,12,13,14,15);
}

DEV v8f wmma_bf16(bf16x16 a, bf16x16 b, v8f c) {
  return __builtin_amdgcn_wmma_f32_16x16x32_bf16(
      false, a, false, b, (short)0, c, false, false);
}

// ---------------------------------------------------------------------------
// fp32 -> bf16 conversion (weights), grid-strided.
// ---------------------------------------------------------------------------
__global__ void cvt_kernel(const float* __restrict__ in, bf16* __restrict__ out,
                           size_t n) {
  size_t i  = (size_t)blockIdx.x * blockDim.x + threadIdx.x;
  size_t st = (size_t)gridDim.x * blockDim.x;
  for (; i < n; i += st) out[i] = (bf16)in[i];
}

// ---------------------------------------------------------------------------
// RMSNorm over D_=1024: one block per token, 256 threads x 4 elems.
// Optional bf16 output (GEMM input) and/or fp32 output.
// ---------------------------------------------------------------------------
__global__ __launch_bounds__(256) void rmsnorm_kernel(
    const float* __restrict__ in, const float* __restrict__ w,
    bf16* __restrict__ outb, float* __restrict__ outf) {
  __shared__ float red[256];
  const int t = threadIdx.x;
  const size_t base = (size_t)blockIdx.x * D_;
  float x[4];
  float s = 0.f;
#pragma unroll
  for (int i = 0; i < 4; ++i) {
    x[i] = in[base + t + i * 256];
    s += x[i] * x[i];
  }
  red[t] = s;
  __syncthreads();
  for (int o = 128; o > 0; o >>= 1) {
    if (t < o) red[t] += red[t + o];
    __syncthreads();
  }
  const float r = rsqrtf(red[0] / (float)D_ + EPS_);
#pragma unroll
  for (int i = 0; i < 4; ++i) {
    const int   c = t + i * 256;
    const float y = x[i] * r * w[c];
    if (outb) outb[base + c] = (bf16)y;
    if (outf) outf[base + c] = y;
  }
}

// ---------------------------------------------------------------------------
// Post-QKV: per-head RMSNorm(q,k) + RoPE + pack q/k bf16 [B,H,L,Dh] and
// v bf16 transposed [B,H,Dh,L]. One block per token, one wave per head.
// ---------------------------------------------------------------------------
__global__ __launch_bounds__(512) void qkv_post_kernel(
    const float* __restrict__ qkv, const int* __restrict__ positions,
    const float* __restrict__ qw, const float* __restrict__ kw,
    bf16* __restrict__ qb, bf16* __restrict__ kb, bf16* __restrict__ vt) {
  const int t    = blockIdx.x;          // token index in [0, B*L)
  const int b    = t >> 10;             // /L_
  const int l    = t & (L_ - 1);
  const int h    = threadIdx.x >> 5;    // wave id == head
  const int lane = threadIdx.x & 31;
  const int d0 = 2 * lane, d1 = 2 * lane + 1;

  const float* row = qkv + (size_t)t * (3 * D_);
  float q0 = row[h * Dh_ + d0],          q1 = row[h * Dh_ + d1];
  float k0 = row[D_ + h * Dh_ + d0],     k1 = row[D_ + h * Dh_ + d1];
  float v0 = row[2 * D_ + h * Dh_ + d0], v1 = row[2 * D_ + h * Dh_ + d1];

  float sq = q0 * q0 + q1 * q1;
  float sk = k0 * k0 + k1 * k1;
#pragma unroll
  for (int m = 1; m < 32; m <<= 1) {
    sq += __shfl_xor(sq, m, 32);
    sk += __shfl_xor(sk, m, 32);
  }
  const float rq = rsqrtf(sq / (float)Dh_ + EPS_);
  const float rk = rsqrtf(sk / (float)Dh_ + EPS_);
  q0 *= rq * qw[d0]; q1 *= rq * qw[d1];
  k0 *= rk * kw[d0]; k1 *= rk * kw[d1];

  if (lane < 16) {  // RoPE on dims [0,32): pair i = lane -> dims (2i, 2i+1)
    const float pos = (float)positions[t];
    const float inv = __expf(-9.210340371976184f * (float)(2 * lane) / 32.f);
    float sn, cs;
    __sincosf(pos * inv, &sn, &cs);
    const float qa = q0 * cs - q1 * sn, qbv = q0 * sn + q1 * cs;
    const float ka = k0 * cs - k1 * sn, kbv = k0 * sn + k1 * cs;
    q0 = qa; q1 = qbv; k0 = ka; k1 = kbv;
  }

  const size_t head = ((size_t)(b * H_ + h)) * L_ * Dh_;
  qb[head + (size_t)l * Dh_ + d0] = (bf16)q0;
  qb[head + (size_t)l * Dh_ + d1] = (bf16)q1;
  kb[head + (size_t)l * Dh_ + d0] = (bf16)k0;
  kb[head + (size_t)l * Dh_ + d1] = (bf16)k1;
  vt[head + (size_t)d0 * L_ + l]  = (bf16)v0;   // transposed [Dh][L]
  vt[head + (size_t)d1 * L_ + l]  = (bf16)v1;
}

// ---------------------------------------------------------------------------
// Flash attention: one wave per (b, h, 16-query tile). Online softmax with
// 16-lane shuffle reductions (wave32 C-layout: lanes 0-15 rows j, 16-31 rows
// j+8). QK^T and P@V via v_wmma_f32_16x16x32_bf16.
// ---------------------------------------------------------------------------
__global__ __launch_bounds__(32) void attn_kernel(
    const bf16* __restrict__ qb, const bf16* __restrict__ kb,
    const bf16* __restrict__ vt, const int* __restrict__ var_id,
    const float* __restrict__ be /* [2][H] for this layer */,
    bf16* __restrict__ ob /* [B,L,H,Dh] */) {
  __shared__ bf16 Pl[16][40];  // P tile, padded

  const int lane = threadIdx.x;
  const int l16 = lane & 15, hif = lane >> 4;
  const int blk = blockIdx.x;
  const int qt = blk & 63;          // L_/16 = 64 tiles
  const int h  = (blk >> 6) & 15;
  const int b  = blk >> 10;
  const int q0 = qt * 16;

  const float scale = 0.125f;  // 1/sqrt(Dh)
  const float be0 = be[h], be1 = be[H_ + h];
  const size_t headQK = ((size_t)(b * H_ + h)) * L_ * Dh_;
  const size_t headV  = headQK;  // same extent, [Dh][L] layout

  // Q A-fragments: two k-steps over Dh=64.
  bf16x16 qa[2];
  {
    const bf16* qp = qb + headQK + (size_t)(q0 + l16) * Dh_ + hif * 8;
    qa[0] = fragA(qp);
    qa[1] = fragA(qp + 32);
  }
  int vq[8];
#pragma unroll
  for (int j = 0; j < 8; ++j) vq[j] = var_id[b * L_ + q0 + hif * 8 + j];

  const v8f vzero = {0.f, 0.f, 0.f, 0.f, 0.f, 0.f, 0.f, 0.f};
  float mrow[8], lrow[8];
  v8f oacc[4];
#pragma unroll
  for (int j = 0; j < 8; ++j) { mrow[j] = -1e30f; lrow[j] = 0.f; }
#pragma unroll
  for (int g = 0; g < 4; ++g) oacc[g] = vzero;

  for (int kt = 0; kt < L_ / 32; ++kt) {
    // ---- S = Q @ K^T for 32 keys (two 16-col n-groups) ----
    v8f s[2];
#pragma unroll
    for (int ng = 0; ng < 2; ++ng) {
      const bf16* kp =
          kb + headQK + (size_t)(kt * 32 + ng * 16 + l16) * Dh_ + hif * 8;
      const bf16x16 k0f = fragB(kp);
      const bf16x16 k1f = fragB(kp + 32);
      v8f z = vzero;
      z = wmma_bf16(qa[0], k0f, z);
      z = wmma_bf16(qa[1], k1f, z);
      s[ng] = z;
    }

    // ---- bias + online softmax ----
    const int vk0 = var_id[b * L_ + kt * 32 + l16];
    const int vk1 = var_id[b * L_ + kt * 32 + 16 + l16];
    float alpha[8];
#pragma unroll
    for (int j = 0; j < 8; ++j) {
      float s0 = s[0][j] * scale + (vq[j] == vk0 ? be1 : be0);
      float s1 = s[1][j] * scale + (vq[j] == vk1 ? be1 : be0);
      float mt = fmaxf(s0, s1);
#pragma unroll
      for (int m = 1; m < 16; m <<= 1) mt = fmaxf(mt, __shfl_xor(mt, m, 32));
      const float mn = fmaxf(mrow[j], mt);
      const float a  = __expf(mrow[j] - mn);
      const float e0 = __expf(s0 - mn);
      const float e1 = __expf(s1 - mn);
      float rs = e0 + e1;
#pragma unroll
      for (int m = 1; m < 16; m <<= 1) rs += __shfl_xor(rs, m, 32);
      lrow[j] = lrow[j] * a + rs;
      mrow[j] = mn;
      alpha[j] = a;
      // store P in C-layout position -> LDS row-major 16x32
      Pl[hif * 8 + j][l16]      = (bf16)e0;
      Pl[hif * 8 + j][16 + l16] = (bf16)e1;
    }
#pragma unroll
    for (int g = 0; g < 4; ++g)
#pragma unroll
      for (int j = 0; j < 8; ++j) oacc[g][j] *= alpha[j];

    // ---- O += P @ V (K-dim = 32 keys, 4 output d-groups) ----
    const bf16x16 pf = fragA(&Pl[l16][hif * 8]);
#pragma unroll
    for (int g = 0; g < 4; ++g) {
      const bf16* vp =
          vt + headV + (size_t)(g * 16 + l16) * L_ + kt * 32 + hif * 8;
      const bf16x16 vf = fragB(vp);
      oacc[g] = wmma_bf16(pf, vf, oacc[g]);
    }
  }

  // ---- normalize and store O as bf16 [B,L,H,Dh] ----
#pragma unroll
  for (int g = 0; g < 4; ++g)
#pragma unroll
    for (int j = 0; j < 8; ++j) {
      const int   row = q0 + hif * 8 + j;
      const float o   = oacc[g][j] / lrow[j];
      ob[(size_t)(b * L_ + row) * D_ + h * Dh_ + g * 16 + l16] = (bf16)o;
    }
}

// ---------------------------------------------------------------------------
// bf16 WMMA GEMM: C[M,N] = A[M,K] @ B[K,N], fp32 accumulate.
// 256 threads = 8 waves in a 2(M) x 4(N) grid; block tile 128x128x32,
// wave tile 64x32 (4 Mfrag x 2 Nfrag). A tile staged via CDNA5 async
// global->LDS (ASYNCcnt) when available; B staged transposed in LDS.
// MODE 0: Cf = acc            (fp32 out)
// MODE 1: Cf = aux + acc      (residual add, fp32 out; aux may alias Cf)
// MODE 2: Cb = silu(aux)*acc  (bf16 out)
// ---------------------------------------------------------------------------
template <int MODE>
__global__ __launch_bounds__(256) void gemm_bf16(
    const bf16* __restrict__ A, const bf16* __restrict__ Bm,
    float* __restrict__ Cf, bf16* __restrict__ Cb,
    const float* __restrict__ aux, int N, int K) {
  __shared__ bf16 As[128][40];
  __shared__ bf16 Bs[128][40];  // transposed: [n][k]

  const int tid = threadIdx.x;
  const int lane = tid & 31, wv = tid >> 5;
  const int l16 = lane & 15, hif = lane >> 4;
  const int wm = wv >> 2, wn = wv & 3;
  const int m0 = blockIdx.y * 128, n0 = blockIdx.x * 128;

  const v8f vzero = {0.f, 0.f, 0.f, 0.f, 0.f, 0.f, 0.f, 0.f};
  v8f acc[4][2];
#pragma unroll
  for (int i = 0; i < 4; ++i)
#pragma unroll
    for (int j = 0; j < 2; ++j) acc[i][j] = vzero;

  const int ar = tid >> 2, ac = (tid & 3) * 8;   // A staging
  const int bk = tid >> 4, bc = (tid & 15) * 8;  // B staging

  for (int k0 = 0; k0 < K; k0 += 32) {
    // stage A tile 128x32 (direct-to-LDS async when the toolchain has it)
#if USE_ASYNC_LDS
#pragma unroll
    for (int r = ar; r < 128; r += 64) {
      __builtin_amdgcn_global_load_async_to_lds_b128(
          (gv4i*)(v4i_*)(A + (size_t)(m0 + r) * K + k0 + ac),
          (lv4i*)(v4i_*)&As[r][ac], 0, 0);
    }
#else
#pragma unroll
    for (int r = ar; r < 128; r += 64) {
      bf16x8 v = *(const bf16x8*)(A + (size_t)(m0 + r) * K + k0 + ac);
      *(bf16x8*)&As[r][ac] = v;
    }
#endif
    // stage B tile 32x128, transposed into LDS [n][k]
#pragma unroll
    for (int k = bk; k < 32; k += 16) {
      bf16x8 v = *(const bf16x8*)(Bm + (size_t)(k0 + k) * N + n0 + bc);
#pragma unroll
      for (int e = 0; e < 8; ++e) Bs[bc + e][k] = v[e];
    }
    if (k0 + 32 < K) {  // L2 prefetch of next tiles (global_prefetch_b8)
      __builtin_prefetch(A + (size_t)(m0 + ar) * K + k0 + 32 + ac, 0, 1);
      __builtin_prefetch(Bm + (size_t)(k0 + 32 + bk) * N + n0 + bc, 0, 1);
    }
#if USE_ASYNC_LDS
    __builtin_amdgcn_s_wait_asynccnt(0);
#endif
    __syncthreads();

    bf16x16 af[4], bfq[2];
#pragma unroll
    for (int mf = 0; mf < 4; ++mf)
      af[mf] = fragA(&As[wm * 64 + mf * 16 + l16][hif * 8]);
#pragma unroll
    for (int nf = 0; nf < 2; ++nf)
      bfq[nf] = fragB(&Bs[wn * 32 + nf * 16 + l16][hif * 8]);
#pragma unroll
    for (int mf = 0; mf < 4; ++mf)
#pragma unroll
      for (int nf = 0; nf < 2; ++nf)
        acc[mf][nf] = wmma_bf16(af[mf], bfq[nf], acc[mf][nf]);
    __syncthreads();
  }

  // epilogue
#pragma unroll
  for (int mf = 0; mf < 4; ++mf) {
#pragma unroll
    for (int nf = 0; nf < 2; ++nf) {
      const int col  = n0 + wn * 32 + nf * 16 + l16;
      const int rowb = m0 + wm * 64 + mf * 16 + hif * 8;
#pragma unroll
      for (int j = 0; j < 8; ++j) {
        const size_t idx = (size_t)(rowb + j) * N + col;
        const float  a   = acc[mf][nf][j];
        if (MODE == 0) {
          Cf[idx] = a;
        } else if (MODE == 1) {
          Cf[idx] = aux[idx] + a;
        } else {
          const float g = aux[idx];
          Cb[idx] = (bf16)(g / (1.f + __expf(-g)) * a);
        }
      }
    }
  }
}

// ---------------------------------------------------------------------------
// Host orchestration.
// ---------------------------------------------------------------------------
extern "C" void kernel_launch(void* const* d_in, const int* in_sizes, int n_in,
                              void* d_out, int out_size, void* d_ws,
                              size_t ws_size, hipStream_t stream) {
  (void)in_sizes; (void)n_in; (void)out_size; (void)ws_size;

  const float* x         = (const float*)d_in[0];
  const int*   var_id    = (const int*)d_in[1];
  const int*   positions = (const int*)d_in[2];
  const float* Wqkv      = (const float*)d_in[3];
  const float* Wo        = (const float*)d_in[4];
  const float* n1w       = (const float*)d_in[5];
  const float* n2w       = (const float*)d_in[6];
  const float* qnw       = (const float*)d_in[7];
  const float* knw       = (const float*)d_in[8];
  const float* bias_emb  = (const float*)d_in[9];
  const float* Wg        = (const float*)d_in[10];
  const float* Wu        = (const float*)d_in[11];
  const float* Wd        = (const float*)d_in[12];
  const float* fnw       = (const float*)d_in[13];

  char*  ws  = (char*)d_ws;
  size_t off = 0;
  auto alloc = [&](size_t bytes) -> void* {
    off = (off + 255) & ~(size_t)255;
    void* p = ws + off;
    off += bytes;
    return p;
  };

  const size_t nWqkv = (size_t)NL_ * D_ * 3 * D_;
  const size_t nWo   = (size_t)NL_ * D_ * D_;
  const size_t nWff  = (size_t)NL_ * D_ * DFF_;

  bf16*  wqkv_b = (bf16*)alloc(nWqkv * 2);
  bf16*  wo_b   = (bf16*)alloc(nWo * 2);
  bf16*  wg_b   = (bf16*)alloc(nWff * 2);
  bf16*  wu_b   = (bf16*)alloc(nWff * 2);
  bf16*  wd_b   = (bf16*)alloc(nWff * 2);
  float* h      = (float*)alloc((size_t)M_ * D_ * 4);
  bf16*  ybf    = (bf16*)alloc((size_t)M_ * D_ * 2);
  float* qkv    = (float*)alloc((size_t)M_ * 3 * D_ * 4);
  bf16*  qbf    = (bf16*)alloc((size_t)M_ * D_ * 2);
  bf16*  kbf    = (bf16*)alloc((size_t)M_ * D_ * 2);
  bf16*  vtb    = (bf16*)alloc((size_t)M_ * D_ * 2);
  bf16*  obf    = (bf16*)alloc((size_t)M_ * D_ * 2);
  float* gate   = (float*)alloc((size_t)M_ * DFF_ * 4);
  bf16*  gub    = (bf16*)alloc((size_t)M_ * DFF_ * 2);

  // weight conversion (once per launch; ~17us at HBM speed)
  cvt_kernel<<<2048, 256, 0, stream>>>(Wqkv, wqkv_b, nWqkv);
  cvt_kernel<<<2048, 256, 0, stream>>>(Wo,   wo_b,   nWo);
  cvt_kernel<<<2048, 256, 0, stream>>>(Wg,   wg_b,   nWff);
  cvt_kernel<<<2048, 256, 0, stream>>>(Wu,   wu_b,   nWff);
  cvt_kernel<<<2048, 256, 0, stream>>>(Wd,   wd_b,   nWff);

  (void)hipMemcpyAsync(h, x, (size_t)M_ * D_ * 4, hipMemcpyDeviceToDevice,
                       stream);

  const dim3 blk256(256);
  for (int l = 0; l < NL_; ++l) {
    const bf16* wq = wqkv_b + (size_t)l * D_ * 3 * D_;
    const bf16* wo = wo_b + (size_t)l * D_ * D_;
    const bf16* wg = wg_b + (size_t)l * D_ * DFF_;
    const bf16* wu = wu_b + (size_t)l * D_ * DFF_;
    const bf16* wd = wd_b + (size_t)l * DFF_ * D_;

    // y = rmsnorm(h, n1)  (bf16 for GEMM)
    rmsnorm_kernel<<<M_, blk256, 0, stream>>>(h, n1w + (size_t)l * D_, ybf,
                                              nullptr);
    // qkv = y @ Wqkv
    gemm_bf16<0><<<dim3(3 * D_ / 128, M_ / 128), blk256, 0, stream>>>(
        ybf, wq, qkv, nullptr, nullptr, 3 * D_, D_);
    // q/k norm + rope + pack
    qkv_post_kernel<<<M_, 512, 0, stream>>>(qkv, positions,
                                            qnw + (size_t)l * Dh_,
                                            knw + (size_t)l * Dh_,
                                            qbf, kbf, vtb);
    // attention
    attn_kernel<<<B_ * H_ * (L_ / 16), 32, 0, stream>>>(
        qbf, kbf, vtb, var_id, bias_emb + (size_t)l * 2 * H_, obf);
    // h += o @ Wo
    gemm_bf16<1><<<dim3(D_ / 128, M_ / 128), blk256, 0, stream>>>(
        obf, wo, h, nullptr, h, D_, D_);
    // y = rmsnorm(h, n2)
    rmsnorm_kernel<<<M_, blk256, 0, stream>>>(h, n2w + (size_t)l * D_, ybf,
                                              nullptr);
    // gate = y @ Wg
    gemm_bf16<0><<<dim3(DFF_ / 128, M_ / 128), blk256, 0, stream>>>(
        ybf, wg, gate, nullptr, nullptr, DFF_, D_);
    // gu = silu(gate) * (y @ Wu)   (bf16)
    gemm_bf16<2><<<dim3(DFF_ / 128, M_ / 128), blk256, 0, stream>>>(
        ybf, wu, nullptr, gub, gate, DFF_, D_);
    // h += gu @ Wd
    gemm_bf16<1><<<dim3(D_ / 128, M_ / 128), blk256, 0, stream>>>(
        gub, wd, h, nullptr, h, D_, DFF_);
  }
  // final rmsnorm -> fp32 output
  rmsnorm_kernel<<<M_, blk256, 0, stream>>>(h, fnw, nullptr, (float*)d_out);
}
